// HypercubeLinear_29446295781761
// MI455X (gfx1250) — matrix operand order repaired and checked by reading, not compiled, not run on hardware
//
#include <hip/hip_runtime.h>
#include <hip/hip_bf16.h>

typedef __attribute__((ext_vector_type(8)))  __bf16 v8bf;
typedef __attribute__((ext_vector_type(16))) __bf16 v16bf;
typedef __attribute__((ext_vector_type(8)))  float  v8f;

#define HDIM    4
#define PNODES  16
#define IN_F    4096
#define OUT_F   4096
#define OB      256
#define IB      256
#define BM      128
#define BN      128
#define BK      32
#define KTOT    ((HDIM + 1) * IB)     // 1280
#define NSTEP   (KTOT / BK)           // 40
#define SROW    40                    // padded LDS row stride in bf16 (80 B, 16B-aligned)
#define TILE_ELEMS (BM * SROW)        // 5120 bf16
#define BUF_ELEMS  (2 * TILE_ELEMS)   // A tile + B tile per buffer
#define W_ELEMS ((long)(HDIM + 1) * PNODES * OB * IB)   // 5242880

// fp32 -> bf16 (RNE), two values packed into one dword.
__device__ __forceinline__ unsigned pk_bf16x2(float a, float b) {
#if __has_builtin(__builtin_amdgcn_cvt_pk_bf16_f32)
  auto r = __builtin_amdgcn_cvt_pk_bf16_f32(a, b);
  union { decltype(r) v; unsigned u; } cv;
  cv.v = r;
  return cv.u;
#else
  unsigned ua = __float_as_uint(a);
  unsigned ub = __float_as_uint(b);
  ua += 0x7FFFu + ((ua >> 16) & 1u);
  ub += 0x7FFFu + ((ub >> 16) & 1u);
  return __builtin_amdgcn_perm(ub, ua, 0x07060302u);  // { ub.hi16, ua.hi16 }
#endif
}

// one-shot fp32 -> bf16 conversion, 8 elements / thread
__global__ __launch_bounds__(256)
void cvt_f32_to_bf16_kernel(const float* __restrict__ src,
                            unsigned short* __restrict__ dst, long n) {
  long i = ((long)blockIdx.x * 256 + threadIdx.x) * 8;
  if (i + 8 > n) return;
  float4 a = *reinterpret_cast<const float4*>(src + i);
  float4 b = *reinterpret_cast<const float4*>(src + i + 4);
  uint4 o;
  o.x = pk_bf16x2(a.x, a.y);
  o.y = pk_bf16x2(a.z, a.w);
  o.z = pk_bf16x2(b.x, b.y);
  o.w = pk_bf16x2(b.z, b.w);
  *reinterpret_cast<uint4*>(dst + i) = o;
}

// Block-sparse hypercube GEMM: out[:, j*256:+256] = sum_b x_blk(i_b) @ W_b^T + bias
// BF16SRC=true : xsrc/wsrc are pre-converted bf16 (from d_ws); tiles staged via
//                GLOBAL_LOAD_ASYNC_TO_LDS_B128 (ASYNCcnt), no VGPR staging.
// BF16SRC=false: xsrc/wsrc are fp32, converted in-register (fallback).
template <bool BF16SRC>
__global__ __launch_bounds__(256)
void HypercubeLinear_wmma_kernel(const void* __restrict__ xsrc,
                                 const void* __restrict__ wsrc,
                                 const float* __restrict__ hb,
                                 float* __restrict__ out,
                                 int M) {
  __shared__ __align__(16) unsigned short smem[2 * BUF_ELEMS]; // 40960 bytes

  const int tid  = threadIdx.x;
  const int lane = tid & 31;
  const int wave = tid >> 5;
  const int wm   = wave >> 1;   // 0..3
  const int wn   = wave & 1;    // 0..1
  const int lr   = lane & 15;
  const int half = lane >> 4;

  const int mtile = blockIdx.x;
  const int ntile = blockIdx.y;
  const int jblk  = blockIdx.z;

  const int m0g = mtile * BM;
  const int n0g = ntile * BN;

  // fp32 staging map: 8 thr/row of 32 floats, 32 rows/pass, 4 passes
  const int c4        = tid & 7;
  const int rowInPass = tid >> 3;

  v8f zero = {0.f, 0.f, 0.f, 0.f, 0.f, 0.f, 0.f, 0.f};
  v8f acc[2][4];
#pragma unroll
  for (int ta = 0; ta < 2; ++ta)
#pragma unroll
    for (int tb = 0; tb < 4; ++tb)
      acc[ta][tb] = zero;

  // ---------------- bf16 path: async-to-LDS staging state ----------------
  // Waves 0-3 carry the A tile (32 rows each), waves 4-7 the B tile.
  // Per wave: 4 issues x 32 lanes x 16B = 2KB per step.
  unsigned          ldsAddrQ[2][4];   // [buffer parity][q]
  unsigned long long gBaseQ[4];       // fixed per-lane global base
  const bool isAwave = (wave < 4);
  if constexpr (BF16SRC) {
    const unsigned ldsStart = (unsigned)(unsigned long long)(&smem[0]);
    const int lw = wave & 3;          // slab index within A (or B)
#pragma unroll
    for (int q = 0; q < 4; ++q) {
      const int chunkid   = q * 32 + lane;     // 0..127
      const int rowInSlab = chunkid >> 2;      // 0..31
      const int chunk     = chunkid & 3;       // 16B chunk within 64B row
      const int row       = lw * 32 + rowInSlab;
      const unsigned tbase = isAwave ? 0u : (unsigned)(TILE_ELEMS * 2);
#pragma unroll
      for (int buf = 0; buf < 2; ++buf)
        ldsAddrQ[buf][q] = ldsStart + (unsigned)(buf * BUF_ELEMS * 2) + tbase +
                           (unsigned)(row * SROW * 2 + chunk * 16);
      if (isAwave)
        gBaseQ[q] = (unsigned long long)xsrc +
                    2ull * ((unsigned long long)(m0g + row) * IN_F + chunk * 8);
      else
        gBaseQ[q] = (unsigned long long)wsrc +
                    2ull * ((unsigned long long)(n0g + row) * IB + chunk * 8);
    }
  }

  auto issueAsync = [&](int s, int buf) {   // bf16 path only
    const int b    = s >> 3;
    const int kin  = (s & 7) * BK;
    const int iblk = (b == 0) ? jblk : (jblk ^ (1 << (b - 1)));
    const unsigned long long soff = isAwave
        ? 2ull * ((unsigned long long)iblk * IB + kin)
        : 2ull * ((unsigned long long)(b * PNODES + iblk) * (OB * IB) + kin);
#pragma unroll
    for (int q = 0; q < 4; ++q) {
      unsigned long long ga = gBaseQ[q] + soff;
      asm volatile("global_load_async_to_lds_b128 %0, %1, off"
                   :: "v"(ldsAddrQ[buf][q]), "v"(ga) : "memory");
    }
  };

  // ---------------- fp32 fallback: register staging ----------------
  float4 ra[4], rb[4];
  auto loadGlobal = [&](int s) {
    const int b    = s >> 3;
    const int kin  = (s & 7) * BK;
    const int iblk = (b == 0) ? jblk : (jblk ^ (1 << (b - 1)));
    const float* xs = (const float*)xsrc + (long)m0g * IN_F + (long)iblk * IB + kin + c4 * 4;
    const float* ws = (const float*)wsrc + ((long)(b * PNODES + iblk) * OB + n0g) * IB + kin + c4 * 4;
#pragma unroll
    for (int p = 0; p < 4; ++p) {
      const int r = p * 32 + rowInPass;
      ra[p] = *reinterpret_cast<const float4*>(xs + (long)r * IN_F);
      rb[p] = *reinterpret_cast<const float4*>(ws + (long)r * IB);
    }
  };
  auto storeLds = [&](int buf) {
    unsigned short* A = smem + buf * BUF_ELEMS;
    unsigned short* B = A + TILE_ELEMS;
#pragma unroll
    for (int p = 0; p < 4; ++p) {
      const int r = p * 32 + rowInPass;
      uint2 pa = make_uint2(pk_bf16x2(ra[p].x, ra[p].y), pk_bf16x2(ra[p].z, ra[p].w));
      uint2 pb = make_uint2(pk_bf16x2(rb[p].x, rb[p].y), pk_bf16x2(rb[p].z, rb[p].w));
      *reinterpret_cast<uint2*>(A + r * SROW + c4 * 4) = pa;
      *reinterpret_cast<uint2*>(B + r * SROW + c4 * 4) = pb;
    }
  };

  auto computeTile = [&](int buf) {
    const unsigned short* A = smem + buf * BUF_ELEMS;
    const unsigned short* B = A + TILE_ELEMS;
    v16bf af[2], bfr[4];
    // A frag (16x32): lanes 0-15 hold K0-7 / K16-23 of row lr; lanes 16-31 K8-15 / K24-31
#pragma unroll
    for (int ta = 0; ta < 2; ++ta) {
      const unsigned short* pa = A + (wm * 32 + ta * 16 + lr) * SROW + half * 8;
      v8bf lo = *reinterpret_cast<const v8bf*>(pa);
      v8bf hi = *reinterpret_cast<const v8bf*>(pa + 16);
      af[ta] = __builtin_shufflevector(lo, hi, 0,1,2,3,4,5,6,7,8,9,10,11,12,13,14,15);
    }
    // B frag (32x16): lane = column; lanes 0-15 K0-7 / K16-23, lanes 16-31 K8-15 / K24-31
#pragma unroll
    for (int tb = 0; tb < 4; ++tb) {
      const unsigned short* pb = B + (wn * 64 + tb * 16 + lr) * SROW + half * 8;
      v8bf lo = *reinterpret_cast<const v8bf*>(pb);
      v8bf hi = *reinterpret_cast<const v8bf*>(pb + 16);
      bfr[tb] = __builtin_shufflevector(lo, hi, 0,1,2,3,4,5,6,7,8,9,10,11,12,13,14,15);
    }
#pragma unroll
    for (int ta = 0; ta < 2; ++ta)
#pragma unroll
      for (int tb = 0; tb < 4; ++tb)
        acc[ta][tb] = __builtin_amdgcn_wmma_f32_16x16x32_bf16(
            false, af[ta], false, bfr[tb], (short)0, acc[ta][tb], false, false);
  };

  // ---------------- pipeline ----------------
  if constexpr (BF16SRC) {
    issueAsync(0, 0);
    asm volatile("s_wait_asynccnt 0x0" ::: "memory");
    __syncthreads();
    for (int s = 0; s < NSTEP; ++s) {
      if (s + 1 < NSTEP) issueAsync(s + 1, (s + 1) & 1);
      computeTile(s & 1);
      if (s + 1 < NSTEP) asm volatile("s_wait_asynccnt 0x0" ::: "memory");
      __syncthreads();
    }
  } else {
    loadGlobal(0);
    storeLds(0);
    __syncthreads();
    for (int s = 0; s < NSTEP; ++s) {
      if (s + 1 < NSTEP) loadGlobal(s + 1);
      computeTile(s & 1);
      if (s + 1 < NSTEP) storeLds((s + 1) & 1);
      __syncthreads();
    }
  }

  // epilogue: C layout (VGPR r, lane L): M = r + 8*(L>=16), N = L&15 ; fuse bias
#pragma unroll
  for (int ta = 0; ta < 2; ++ta) {
#pragma unroll
    for (int tb = 0; tb < 4; ++tb) {
      const int ng   = jblk * OB + n0g + wn * 64 + tb * 16 + lr;
      const float bv = hb[ng];
      const int mg   = m0g + wm * 32 + ta * 16 + half * 8;
      float* dst = out + (long)mg * OUT_F + ng;
#pragma unroll
      for (int r = 0; r < 8; ++r)
        dst[(long)r * OUT_F] = acc[ta][tb][r] + bv;
    }
  }
}

extern "C" void kernel_launch(void* const* d_in, const int* in_sizes, int n_in,
                              void* d_out, int out_size, void* d_ws, size_t ws_size,
                              hipStream_t stream) {
  (void)n_in; (void)out_size;
  const float* x  = (const float*)d_in[0];   // [M, 4096] fp32
  const float* hw = (const float*)d_in[1];   // [5, 16, 256, 256] fp32
  const float* hb = (const float*)d_in[2];   // [4096] fp32
  float* out = (float*)d_out;

  const int  M      = in_sizes[0] / IN_F;    // 8192
  const long xElems = (long)M * IN_F;
  const size_t needWs = (size_t)(xElems + W_ELEMS) * sizeof(unsigned short);

  dim3 grid(M / BM, OB / BN, PNODES);
  dim3 block(256);

  if (d_ws && ws_size >= needWs) {
    unsigned short* xbf = (unsigned short*)d_ws;
    unsigned short* wbf = xbf + xElems;
    cvt_f32_to_bf16_kernel<<<(unsigned)(xElems / (256 * 8)), 256, 0, stream>>>(x, xbf, xElems);
    cvt_f32_to_bf16_kernel<<<(unsigned)(W_ELEMS / (256 * 8)), 256, 0, stream>>>(hw, wbf, W_ELEMS);
    HypercubeLinear_wmma_kernel<true><<<grid, block, 0, stream>>>(xbf, wbf, hb, out, M);
  } else {
    HypercubeLinear_wmma_kernel<false><<<grid, block, 0, stream>>>(x, hw, hb, out, M);
  }
}
